// VectorQuantizer_66245575573921
// MI455X (gfx1250) — matrix operand order repaired and checked by reading, not compiled, not run on hardware
//
#include <hip/hip_runtime.h>
#include <math.h>

typedef __attribute__((ext_vector_type(16))) __bf16 v16bf;
typedef __attribute__((ext_vector_type(8)))  float  v8f;

#define EDIM   256
#define HW_    1024
#define NTOK   8192
#define NE     8192
#define NBATCH 8
#define BETA_  0.25f
#define NSPLIT 4          // codebook split factor (8192 codes -> 4 x 2048)
#define KT_PER_SPLIT 128  // 512 code tiles / NSPLIT

// ---- workspace byte offsets (total ~16.4 MB) ----
#define WS_A_HI  (0u)                      // 8192 tok x 256 feat bf16 (WMMA A layout)  4MB
#define WS_A_LO  (4u<<20)                  // low-residual split                        4MB
#define WS_B_HI  (8u<<20)                  // 8192 codes x 256 feat bf16 (WMMA B)       4MB
#define WS_B_LO  (12u<<20)                 //                                           4MB
#define WS_ENORM (16u<<20)                 // ||e_k||^2 fp32, 8192                      32KB
#define WS_IDX   ((16u<<20)+32768u)        // argmin idx i32, 8192                      32KB
#define WS_CNT   ((16u<<20)+65536u)        // histogram u32, 8192                       32KB
#define WS_PSUM  ((16u<<20)+98304u)        // per-token sq-err fp32, 8192               32KB
#define WS_PBEST ((16u<<20)+131072u)       // partial best f32, NSPLIT x 8192           128KB
#define WS_PIDX  ((16u<<20)+262144u)       // partial idx i32,  NSPLIT x 8192           128KB

// ---- d_out element offsets (fp32, concatenated in reference return order) ----
#define OUT_ZQ     0         // [8,256,32,32] = 2097152
#define OUT_LOSS   2097152   // [8]
#define OUT_IDX    2097160   // [8192] (as float)
#define OUT_COMMIT 2105352   // scalar
#define OUT_EMB    2105353   // scalar
#define OUT_PERP   2105354   // scalar

__device__ __forceinline__ void bf16_split(float x, unsigned short& h, unsigned short& l) {
    __bf16 hb = (__bf16)x;            // RNE convert
    float  hf = (float)hb;
    __bf16 lb = (__bf16)(x - hf);     // residual
    __builtin_memcpy(&h, &hb, 2);
    __builtin_memcpy(&l, &lb, 2);
}

// Zero histogram bins.
__global__ __launch_bounds__(256) void vq_init(unsigned* __restrict__ counts) {
    int i = blockIdx.x * 256 + threadIdx.x;
    if (i < NE) counts[i] = 0u;
}

// Pack z (NCHW) -> WMMA A-operand layout, bf16 hi/lo.
// A 16-bit 16x32 layout: lanes 0-15 (row M=lane): V0-3 = K0..7, V4-7 = K16..23;
//                        lanes 16-31 (row M=lane-16): V0-3 = K8..15, V4-7 = K24..31.
__global__ __launch_bounds__(256) void vq_pack_z(const float* __restrict__ z,
                                                 unsigned short* __restrict__ Ahi,
                                                 unsigned short* __restrict__ Alo) {
    int n = blockIdx.x;          // token 0..8191  (n = b*1024 + hw)
    int f = threadIdx.x;         // feature 0..255
    float x = z[(n >> 10) * (EDIM * HW_) + f * HW_ + (n & 1023)];
    unsigned short h, l;
    bf16_split(x, h, l);
    int mt = n >> 4, m = n & 15, q = f >> 5, kk = f & 31;
    int sub = kk & 15, grp = kk >> 4;
    int lane = (sub < 8) ? m : (16 + m);
    int v = ((sub & 7) >> 1) + grp * 4;
    int dst = ((((mt * 8 + q) * 32 + lane) * 8 + v) << 1) + (kk & 1);
    Ahi[dst] = h;
    Alo[dst] = l;
}

// Pack emb -> WMMA B-operand layout (B is KxN=32x16 per chunk), bf16 hi/lo; also ||e||^2.
// B 16-bit layout: lanes 0-15: N=lane, V0-3 = K0..7, V4-7 = K16..23; lanes 16-31: N=lane-16, V0-3 = K8..15, V4-7 = K24..31.
__global__ __launch_bounds__(256) void vq_pack_e(const float* __restrict__ emb,
                                                 unsigned short* __restrict__ Bhi,
                                                 unsigned short* __restrict__ Blo,
                                                 float* __restrict__ enorm) {
    __shared__ float red[256];
    int k = blockIdx.x;          // code 0..8191
    int f = threadIdx.x;         // feature 0..255
    float x = emb[k * EDIM + f];
    unsigned short h, l;
    bf16_split(x, h, l);
    int kt = k >> 4, ncol = k & 15, q = f >> 5, kk = f & 31;
    int lane = ((kk & 15) < 8) ? ncol : (16 + ncol);
    int v = ((kk & 7) >> 1) + (kk >> 4) * 4;
    int dst = ((((kt * 8 + q) * 32 + lane) * 8 + v) << 1) + (kk & 1);
    Bhi[dst] = h;
    Blo[dst] = l;
    red[f] = x * x;
    __syncthreads();
    for (int s = 128; s > 0; s >>= 1) { if (f < s) red[f] += red[f + s]; __syncthreads(); }
    if (f == 0) enorm[k] = red[0];
}

// Fused distance GEMM + partial argmin over one codebook quarter.
// Grid: 256 WGs x 8 waves = 2048 waves. WG g: token-tile group = g>>2, code split = g&3.
// Wave w handles token tile mt = (g>>2)*8 + w over code tiles [split*128, split*128+128).
// score = ||e||^2 - 2*(zhi*ehi + zhi*elo + zlo*ehi)   (||z||^2 constant per token, dropped)
// Three independent accumulators break the WMMA D->C dependency chain so consecutive
// v_wmma ops are independent and the matrix pipe issues back-to-back.
__global__ __launch_bounds__(256) void vq_gemm_argmin(const uint4* __restrict__ Ahi,
                                                      const uint4* __restrict__ Alo,
                                                      const uint4* __restrict__ Bhi,
                                                      const uint4* __restrict__ Blo,
                                                      const float* __restrict__ enorm,
                                                      float* __restrict__ pbest,
                                                      int* __restrict__ pidx) {
    __shared__ float sbest[256][8];
    __shared__ int   sidx[256][8];
    const int tid   = threadIdx.x;
    const int wave  = tid >> 5;
    const int lane  = tid & 31;
    const int split = blockIdx.x & (NSPLIT - 1);
    const int mt    = (blockIdx.x >> 2) * 8 + wave;
    const int col   = lane & 31 & 15;
    const int kt0   = split * KT_PER_SPLIT;

    union BV { uint4 q[2]; v16bf v; };

    // Preload this wave's 16-token A tile (hi/lo), 8 K-chunks each.
    v16bf ahi[8], alo[8];
#pragma unroll
    for (int q = 0; q < 8; ++q) {
        BV t;
        const uint4* p = Ahi + (((mt * 8 + q) * 32 + lane) * 2);
        t.q[0] = p[0]; t.q[1] = p[1];
        ahi[q] = t.v;
        const uint4* p2 = Alo + (((mt * 8 + q) * 32 + lane) * 2);
        t.q[0] = p2[0]; t.q[1] = p2[1];
        alo[q] = t.v;
    }

    float best[8];
    int   besti[8];
#pragma unroll
    for (int d = 0; d < 8; ++d) { best[d] = 3.4e38f; besti[d] = 0; }

    for (int kt = kt0; kt < kt0 + KT_PER_SPLIT; ++kt) {
        // Load B tile (16 codes x 256 feats), hi/lo, fully into registers.
        v16bf bhi[8], blo[8];
#pragma unroll
        for (int q = 0; q < 8; ++q) {
            BV t;
            const uint4* p = Bhi + (((kt * 8 + q) * 32 + lane) * 2);
            t.q[0] = p[0]; t.q[1] = p[1];
            bhi[q] = t.v;
            const uint4* p2 = Blo + (((kt * 8 + q) * 32 + lane) * 2);
            t.q[0] = p2[0]; t.q[1] = p2[1];
            blo[q] = t.v;
        }
        float el = enorm[kt * 16 + col];
        if (kt + 1 < kt0 + KT_PER_SPLIT) {
            __builtin_prefetch((const void*)(Bhi + ((((kt + 1) * 8) * 32 + lane) * 2)), 0, 0);
            __builtin_prefetch((const void*)(Blo + ((((kt + 1) * 8) * 32 + lane) * 2)), 0, 0);
        }

        v8f c0 = {}, c1 = {}, c2 = {};     // independent chains: hi*hi, hi*lo, lo*hi
#pragma unroll
        for (int q = 0; q < 8; ++q) {
            c0 = __builtin_amdgcn_wmma_f32_16x16x32_bf16(false, ahi[q], false, bhi[q], (short)0, c0, false, false);
            c1 = __builtin_amdgcn_wmma_f32_16x16x32_bf16(false, ahi[q], false, blo[q], (short)0, c1, false, false);
            c2 = __builtin_amdgcn_wmma_f32_16x16x32_bf16(false, alo[q], false, bhi[q], (short)0, c2, false, false);
        }

        const int code = kt * 16 + col;
#pragma unroll
        for (int d = 0; d < 8; ++d) {
            float s = el - 2.0f * (c0[d] + c1[d] + c2[d]);
            if (s < best[d]) { best[d] = s; besti[d] = code; }   // strict < keeps first min
        }
    }

    // Cross-lane argmin: C/D layout is VGPR d -> row d (lanes 0-15) / row 8+d (lanes 16-31).
#pragma unroll
    for (int d = 0; d < 8; ++d) { sbest[tid][d] = best[d]; sidx[tid][d] = besti[d]; }
    __syncthreads();
    if (lane < 16) {
        int t = lane;                               // token within tile
        int srcBase = wave * 32 + ((t < 8) ? 0 : 16);
        int d = t & 7;
        float bv = 3.4e38f;
        int   bi = 0x7fffffff;
        for (int j = 0; j < 16; ++j) {
            float v = sbest[srcBase + j][d];
            int   i = sidx[srcBase + j][d];
            if (v < bv || (v == bv && i < bi)) { bv = v; bi = i; }
        }
        int n = mt * 16 + t;
        pbest[split * NTOK + n] = bv;
        pidx [split * NTOK + n] = bi;
    }
}

// Merge the NSPLIT partial argmins per token (splits are in ascending code order,
// so strict < preserves first-occurrence semantics of jnp.argmin).
__global__ __launch_bounds__(256) void vq_merge(const float* __restrict__ pbest,
                                                const int* __restrict__ pidx,
                                                int* __restrict__ idx_out,
                                                float* __restrict__ idxf_out) {
    int n = blockIdx.x * 256 + threadIdx.x;
    if (n >= NTOK) return;
    float bv = 3.4e38f;
    int   bi = 0x7fffffff;
#pragma unroll
    for (int s = 0; s < NSPLIT; ++s) {
        float v = pbest[s * NTOK + n];
        int   i = pidx [s * NTOK + n];
        if (v < bv) { bv = v; bi = i; }
    }
    idx_out[n]  = bi;
    idxf_out[n] = (float)bi;
}

// Gather z_q = emb[idx], write NCHW output, per-token squared error, histogram.
__global__ __launch_bounds__(256) void vq_gather_loss(const float* __restrict__ z,
                                                      const float* __restrict__ emb,
                                                      const int* __restrict__ idx,
                                                      float* __restrict__ zq_out,
                                                      float* __restrict__ persum,
                                                      unsigned* __restrict__ counts) {
    __shared__ float red[256];
    int n = blockIdx.x, c = threadIdx.x;
    int k = idx[n];
    int b = n >> 10, hw = n & 1023;
    float e  = emb[k * EDIM + c];
    int  off = b * (EDIM * HW_) + c * HW_ + hw;
    float zv = z[off];
    zq_out[off] = e;                 // z_q_st == z_q numerically
    float dlt = e - zv;
    red[c] = dlt * dlt;
    __syncthreads();
    for (int s = 128; s > 0; s >>= 1) { if (c < s) red[c] += red[c + s]; __syncthreads(); }
    if (c == 0) {
        persum[n] = red[0];          // deterministic (no float atomics)
        atomicAdd(&counts[k], 1u);
    }
}

// Per-batch losses, scalar losses, perplexity.
__global__ __launch_bounds__(256) void vq_finalize(const float* __restrict__ persum,
                                                   const unsigned* __restrict__ counts,
                                                   float* __restrict__ out) {
    __shared__ float red[256];
    int t = threadIdx.x;
    float total = 0.0f;
    for (int b = 0; b < NBATCH; ++b) {
        float s = 0.0f;
        for (int j = t; j < HW_; j += 256) s += persum[b * HW_ + j];
        red[t] = s;
        __syncthreads();
        for (int st = 128; st > 0; st >>= 1) { if (t < st) red[t] += red[t + st]; __syncthreads(); }
        if (t == 0) {
            float mse = red[0] * (1.0f / 262144.0f);   // mean over C*H*W
            out[OUT_LOSS + b] = (1.0f + BETA_) * mse;  // emb_loss + beta*commit_loss
            total += red[0];
        }
        __syncthreads();
    }
    float h = 0.0f;
    for (int k = t; k < NE; k += 256) {
        float p = (float)counts[k] * (1.0f / (float)NTOK);
        h += p * logf(p + 1e-10f);
    }
    red[t] = h;
    __syncthreads();
    for (int st = 128; st > 0; st >>= 1) { if (t < st) red[t] += red[t + st]; __syncthreads(); }
    if (t == 0) {
        float mse_all = total * (1.0f / (8.0f * 262144.0f));
        out[OUT_COMMIT] = mse_all;           // mean(commit_loss)
        out[OUT_EMB]    = BETA_ * mse_all;   // beta * mean(emb_loss)
        out[OUT_PERP]   = expf(-red[0]);
    }
}

extern "C" void kernel_launch(void* const* d_in, const int* in_sizes, int n_in,
                              void* d_out, int out_size, void* d_ws, size_t ws_size,
                              hipStream_t stream) {
    (void)in_sizes; (void)n_in; (void)out_size; (void)ws_size;
    const float* z   = (const float*)d_in[0];   // [8,256,32,32]
    const float* emb = (const float*)d_in[1];   // [8192,256]
    float* out = (float*)d_out;
    char*  ws  = (char*)d_ws;

    unsigned short* Ahi   = (unsigned short*)(ws + WS_A_HI);
    unsigned short* Alo   = (unsigned short*)(ws + WS_A_LO);
    unsigned short* Bhi   = (unsigned short*)(ws + WS_B_HI);
    unsigned short* Blo   = (unsigned short*)(ws + WS_B_LO);
    float*          enorm = (float*)(ws + WS_ENORM);
    int*            idxw  = (int*)(ws + WS_IDX);
    unsigned*       cnt   = (unsigned*)(ws + WS_CNT);
    float*          psum  = (float*)(ws + WS_PSUM);
    float*          pbest = (float*)(ws + WS_PBEST);
    int*            pidx  = (int*)(ws + WS_PIDX);

    vq_init<<<(NE + 255) / 256, 256, 0, stream>>>(cnt);
    vq_pack_z<<<NTOK, 256, 0, stream>>>(z, Ahi, Alo);
    vq_pack_e<<<NE, 256, 0, stream>>>(emb, Bhi, Blo, enorm);
    vq_gemm_argmin<<<64 * NSPLIT, 256, 0, stream>>>((const uint4*)Ahi, (const uint4*)Alo,
                                                    (const uint4*)Bhi, (const uint4*)Blo,
                                                    enorm, pbest, pidx);
    vq_merge<<<(NTOK + 255) / 256, 256, 0, stream>>>(pbest, pidx, idxw, out + OUT_IDX);
    vq_gather_loss<<<NTOK, 256, 0, stream>>>(z, emb, idxw, out + OUT_ZQ, psum, cnt);
    vq_finalize<<<1, 256, 0, stream>>>(psum, cnt, out);
}
